// StateSpaceModel_75557064671981
// MI455X (gfx1250) — matrix operand order repaired and checked by reading, not compiled, not run on hardware
//
#include <hip/hip_runtime.h>

// ---------------------------------------------------------------------------
// Problem constants (from reference)
// ---------------------------------------------------------------------------
#define T_STEPS 128
#define BATCH   256
#define ADIM    16
#define ZDIM    32
#define KMIX    8
#define HID     128
#define G4      (4 * HID)       // 512 gates
#define BT      16              // batch tile per workgroup
#define NWG     (BATCH / BT)    // 16 workgroups
#define NTHREADS 512            // 16 waves (wave32)
#define K0      160             // [a(16) | h0(128) | pad(16)]
#define K1      256             // [h0(128) | h1(128)]
#define KC0     5               // K-chunks of 32 for layer 0
#define KC1     8               // K-chunks of 32 for layer 1
#define NTILES  (G4 / 16)       // 32 gate tiles of 16
#define SCR     3072            // per-batch scratch floats

// ---------------------------------------------------------------------------
// LDS layout (dynamic shared memory, byte offsets)
// ---------------------------------------------------------------------------
// per-batch scratch (stride SCR floats):
//   [   0,1024) Abuf (A_next)   | gates[0,512) during LSTM phase
//   [1024,1536) Cbuf            \  covt aliases [1024,2048)
//   [1536,2048) Saug (16x32)    /
//   [2048,2560) CP (16x32)      \  M1 aliases [2048,3072)
//   [2560,3072) Kt (32x16)      /
constexpr int OFF_SCRATCH = 0;
constexpr int OFF_C0    = OFF_SCRATCH + BT * SCR * 4;        // 196608
constexpr int OFF_C1    = OFF_C0    + BT * HID * 4;
constexpr int OFF_COVP  = OFF_C1    + BT * HID * 4;
constexpr int OFF_MEANP = OFF_COVP  + BT * ZDIM * ZDIM * 4;
constexpr int OFF_MEANT = OFF_MEANP + BT * ZDIM * 4;
constexpr int OFF_INNOV = OFF_MEANT + BT * ZDIM * 4;
constexpr int OFF_ATILE = OFF_INNOV + BT * ADIM * 4;
constexpr int OFF_WMIX  = OFF_ATILE + BT * ADIM * 4;
constexpr int OFF_LOGIT = OFF_WMIX  + BT * KMIX * 4;
constexpr int OFF_X0    = OFF_LOGIT + BT * KMIX * 4;
constexpr int OFF_X1    = OFF_X0    + BT * K0 * 2;
constexpr int SMEM_BYTES = OFF_X1   + BT * K1 * 2;           // 299008 B < 320KB

// workspace layout (bytes)
constexpr size_t WS_W0P = 0;                                  // 32*5*512 bf16
constexpr size_t WS_W1P = WS_W0P + (size_t)NTILES * KC0 * 512 * 2;   // 163840
constexpr size_t WS_B0S = WS_W1P + (size_t)NTILES * KC1 * 512 * 2;   // 425984
constexpr size_t WS_B1S = WS_B0S + G4 * 4;
constexpr size_t WS_QM  = WS_B1S + G4 * 4;
constexpr size_t WS_RM  = WS_QM  + ZDIM * ZDIM * 4;

// ---------------------------------------------------------------------------
// WMMA types / helpers
// ---------------------------------------------------------------------------
typedef __bf16 bf16x16 __attribute__((ext_vector_type(16)));
typedef float  f32x8   __attribute__((ext_vector_type(8)));
// Plain builtin vector (NOT HIP's class-type uint4) so loads through an
// address-space(1) pointer are legal builtin-vector loads on host & device.
typedef unsigned int u32x4 __attribute__((ext_vector_type(4)));

// Global (addrspace 1) vector pointer: forces global_load_b128 (LOADcnt only)
// instead of flat_load_b128 (which also ties up DScnt / LDS path).
typedef __attribute__((address_space(1))) const u32x4* gp4;

union FragAB { bf16x16 v; u32x4 u[2]; };

static __device__ __forceinline__ unsigned short f2bf(float f) {
  union { float f; unsigned u; } x; x.f = f;
  unsigned r = x.u + 0x7FFFu + ((x.u >> 16) & 1u);   // RNE
  return (unsigned short)(r >> 16);
}
static __device__ __forceinline__ float bf2f(unsigned short b) {
  union { unsigned u; float f; } x; x.u = ((unsigned)b) << 16; return x.f;
}
static __device__ __forceinline__ float sigmoidf_(float x) {
  return 1.f / (1.f + __expf(-x));
}

// One wave computes a 16(batch) x 32(gate) slab of the gate GEMM with
// v_wmma_f32_16x16x32_bf16, accumulating over KC chunks of K=32.
// A-fragments come from LDS (bf16 activations), B-fragments from global
// pre-packed weights (contiguous 1KB per fragment, lane-major).
template <int KC>
static __device__ __forceinline__ void gemm_tile(
    const unsigned short* __restrict__ xcat, int kstride,
    const unsigned short* __restrict__ wpack,
    const float* __restrict__ bsum,
    float* __restrict__ scratch, int lane, int wv)
{
  // Defeat LICM: without this, all KC*2 weight fragments get hoisted out of
  // the 128-step time loop, blow the register budget and spill to scratch.
  // The opaque pointer keeps them as per-iteration loads (L2 hits).
  const unsigned short* wp = wpack;
  asm volatile("" : "+s"(wp));

  const int nt0 = wv * 2, nt1 = wv * 2 + 1;
  const int m   = lane & 15;          // batch row for A
  const int sel = lane >> 4;          // K-half select (contiguous k order, k = 16*sel + s)
  f32x8 acc0 = {0.f,0.f,0.f,0.f,0.f,0.f,0.f,0.f};
  f32x8 acc1 = {0.f,0.f,0.f,0.f,0.f,0.f,0.f,0.f};
  const unsigned short* arow = xcat + m * kstride;
#pragma unroll
  for (int kc = 0; kc < KC; ++kc) {
    FragAB fa, fb0, fb1;
    // A: lanes<16 get K = kb+[0..15]; lanes>=16 get kb+[16..31]
    fa.u[0] = *(const u32x4*)(arow + kc * 32 + sel * 16);
    fa.u[1] = *(const u32x4*)(arow + kc * 32 + sel * 16 + 8);
    // B: packed fragments, 512 bf16 each, lane-major (16 bf16 per lane).
    // Integer round-trip re-establishes the global address space so these
    // lower to global_load_b128 rather than flat_load_b128.
    gp4 bp0 = (gp4)(unsigned long long)(wp + ((size_t)(nt0 * KC + kc) << 9)) + (lane << 1);
    gp4 bp1 = (gp4)(unsigned long long)(wp + ((size_t)(nt1 * KC + kc) << 9)) + (lane << 1);
    fb0.u[0] = bp0[0]; fb0.u[1] = bp0[1];
    fb1.u[0] = bp1[0]; fb1.u[1] = bp1[1];
    acc0 = __builtin_amdgcn_wmma_f32_16x16x32_bf16(false, fa.v, false, fb0.v,
                                                   (short)0, acc0, false, false);
    acc1 = __builtin_amdgcn_wmma_f32_16x16x32_bf16(false, fa.v, false, fb1.v,
                                                   (short)0, acc1, false, false);
  }
  // D layout: lanes 0-15: N=lane, M=r ; lanes 16-31: N=lane-16, M=8+r
  const int n   = lane & 15;
  const int mhi = (lane >> 4) * 8;
  const int g0 = nt0 * 16 + n, g1 = nt1 * 16 + n;
  const float bb0 = bsum[g0], bb1 = bsum[g1];
#pragma unroll
  for (int r = 0; r < 8; ++r) {
    int mm = r + mhi;                              // batch row
    scratch[mm * SCR + g0] = acc0[r] + bb0;        // gates live at scr+[0,512)
    scratch[mm * SCR + g1] = acc1[r] + bb1;
  }
}

// ---------------------------------------------------------------------------
// Prep: pack weights into WMMA B-fragments (bf16), bias sums, Q and R.
// ---------------------------------------------------------------------------
__global__ void ssm_prep(const float* __restrict__ Wih0, const float* __restrict__ Whh0,
                         const float* __restrict__ bih0, const float* __restrict__ bhh0,
                         const float* __restrict__ Wih1, const float* __restrict__ Whh1,
                         const float* __restrict__ bih1, const float* __restrict__ bhh1,
                         const float* __restrict__ QL,   const float* __restrict__ RL,
                         unsigned short* __restrict__ w0p, unsigned short* __restrict__ w1p,
                         float* __restrict__ b0s, float* __restrict__ b1s,
                         float* __restrict__ Qm,  float* __restrict__ Rm)
{
  const int N0 = NTILES * KC0 * 512;   // 81920
  const int N1 = NTILES * KC1 * 512;   // 131072
  int tid = blockIdx.x * blockDim.x + threadIdx.x;
  if (tid < N0) {
    int frag = tid >> 9, w = tid & 511;
    int nt = frag / KC0, kc = frag % KC0;
    int lane = w >> 4, pos = w & 15;
    int k = kc * 32 + ((lane >> 4) << 4) + pos;    // B: lane<16 -> K 0..15 of chunk
    int gate = nt * 16 + (lane & 15);
    float v = 0.f;
    if (k < ADIM)            v = Wih0[gate * ADIM + k];
    else if (k < ADIM + HID) v = Whh0[gate * HID + (k - ADIM)];
    w0p[tid] = f2bf(v);
  } else if (tid < N0 + N1) {
    int t2 = tid - N0;
    int frag = t2 >> 9, w = t2 & 511;
    int nt = frag / KC1, kc = frag % KC1;
    int lane = w >> 4, pos = w & 15;
    int k = kc * 32 + ((lane >> 4) << 4) + pos;
    int gate = nt * 16 + (lane & 15);
    float v = (k < HID) ? Wih1[gate * HID + k] : Whh1[gate * HID + (k - HID)];
    w1p[t2] = f2bf(v);
  } else if (tid < N0 + N1 + G4) {
    int g = tid - (N0 + N1);
    b0s[g] = bih0[g] + bhh0[g];
  } else if (tid < N0 + N1 + 2 * G4) {
    int g = tid - (N0 + N1 + G4);
    b1s[g] = bih1[g] + bhh1[g];
  } else if (tid < N0 + N1 + 2 * G4 + ZDIM * ZDIM) {
    int e = tid - (N0 + N1 + 2 * G4);
    int i = e >> 5, j = e & 31;
    float s = 0.f;
    for (int k = 0; k < ZDIM; ++k) s += QL[i * ZDIM + k] * QL[j * ZDIM + k];
    Qm[e] = s + ((i == j) ? 0.001f : 0.f);
  } else if (tid < N0 + N1 + 2 * G4 + ZDIM * ZDIM + ADIM * ADIM) {
    int e = tid - (N0 + N1 + 2 * G4 + ZDIM * ZDIM);
    int i = e >> 4, j = e & 15;
    float s = 0.f;
    for (int k = 0; k < ADIM; ++k) s += RL[i * ADIM + k] * RL[j * ADIM + k];
    Rm[e] = s + ((i == j) ? 0.001f : 0.f);
  }
}

// ---------------------------------------------------------------------------
// Main: full T-step scan; one WG owns 16 batches, state resident in LDS.
// ---------------------------------------------------------------------------
__global__ __launch_bounds__(NTHREADS, 1)
void ssm_main(const float* __restrict__ as_,
              const float* __restrict__ AK, const float* __restrict__ CK,
              const float* __restrict__ init_mean, const float* __restrict__ init_cov,
              const float* __restrict__ Wout, const float* __restrict__ bout,
              const unsigned short* __restrict__ w0p, const unsigned short* __restrict__ w1p,
              const float* __restrict__ b0s, const float* __restrict__ b1s,
              const float* __restrict__ Qm,  const float* __restrict__ Rm,
              float* __restrict__ out)
{
  extern __shared__ __align__(16) char smem[];
  float* scratch = (float*)(smem + OFF_SCRATCH);
  float* c0s   = (float*)(smem + OFF_C0);
  float* c1s   = (float*)(smem + OFF_C1);
  float* covps = (float*)(smem + OFF_COVP);
  float* meanps= (float*)(smem + OFF_MEANP);
  float* meants= (float*)(smem + OFF_MEANT);
  float* innovs= (float*)(smem + OFF_INNOV);
  float* atiles= (float*)(smem + OFF_ATILE);
  float* wmixs = (float*)(smem + OFF_WMIX);
  float* logitss = (float*)(smem + OFF_LOGIT);
  unsigned short* xcat0 = (unsigned short*)(smem + OFF_X0);
  unsigned short* xcat1 = (unsigned short*)(smem + OFF_X1);

  const int tid  = threadIdx.x;
  const int lane = tid & 31;
  const int wv   = tid >> 5;
  const int Bbase = blockIdx.x * BT;

  const size_t o_means = 0;
  const size_t o_covs  = o_means + (size_t)T_STEPS * BATCH * ZDIM;
  const size_t o_nmean = o_covs  + (size_t)T_STEPS * BATCH * ZDIM * ZDIM;
  const size_t o_ncov  = o_nmean + (size_t)T_STEPS * BATCH * ZDIM;
  const size_t o_matA  = o_ncov  + (size_t)T_STEPS * BATCH * ZDIM * ZDIM;
  const size_t o_matC  = o_matA  + (size_t)(T_STEPS + 1) * BATCH * ZDIM * ZDIM;
  const size_t o_aout  = o_matC  + (size_t)(T_STEPS + 1) * BATCH * ADIM * ZDIM;

  // ---- init persistent state ----
  for (int e = tid; e < BT * HID; e += NTHREADS) { c0s[e] = 0.f; c1s[e] = 0.f; }
  for (int e = tid; e < BT * K0; e += NTHREADS)  xcat0[e] = 0;
  for (int e = tid; e < BT * K1; e += NTHREADS)  xcat1[e] = 0;
  for (int e = tid; e < BT * KMIX; e += NTHREADS) wmixs[e] = 1.f;  // carry w0 = ones
  for (int e = tid; e < BT * ZDIM; e += NTHREADS) meanps[e] = init_mean[e & (ZDIM - 1)];
  for (int e = tid; e < BT * ZDIM * ZDIM; e += NTHREADS)
    covps[e] = init_cov[e & (ZDIM * ZDIM - 1)];
  __syncthreads();

  const int b = wv;        // batch lane-group mapping: wave == batch
  const int i = lane;      // matrix row 0..31
  float* scr  = scratch + b * SCR;
  float* Cb   = scr + 1024;
  float* Saug = scr + 1536;
  float* CP   = scr + 2048;
  float* Kt   = scr + 2560;
  float* covt = scr + 1024;   // alias Cb+Saug (dead by then)
  float* M1   = scr + 2048;   // alias CP+Kt  (dead by then)
  float* Ab   = scr;          // A_next [0,1024); gates alias [0,512) in LSTM

  for (int t = 0; t < T_STEPS; ++t) {
    const size_t brow = (size_t)t * BATCH + Bbase + b;
    // ---- load a_t tile, emit a_out ----
    if (i < ADIM) {
      float av = as_[brow * ADIM + i];
      atiles[b * ADIM + i] = av;
      out[o_aout + brow * ADIM + i] = av;
    }
    // ---- mixture matrices from current w: stream A(w), store C(w) ----
    {
      float wr[KMIX];
#pragma unroll
      for (int k = 0; k < KMIX; ++k) wr[k] = wmixs[b * KMIX + k];
      float* oA = out + o_matA + brow * ZDIM * ZDIM + i * ZDIM;
      for (int j = 0; j < ZDIM; ++j) {
        float s = 0.f;
#pragma unroll
        for (int k = 0; k < KMIX; ++k) s += wr[k] * AK[(k * ZDIM + i) * ZDIM + j];
        oA[j] = s;
      }
      if (i < ADIM) {
        float* oC = out + o_matC + brow * ADIM * ZDIM + i * ZDIM;
        for (int j = 0; j < ZDIM; ++j) {
          float s = 0.f;
#pragma unroll
          for (int k = 0; k < KMIX; ++k) s += wr[k] * CK[(k * ADIM + i) * ZDIM + j];
          Cb[i * ZDIM + j] = s; oC[j] = s;
        }
      }
    }
    __syncthreads();
    // ---- CP = C @ cov_p (16x32) ----
    {
      int m = i & 15, j0 = (i >> 4) * 16;
      for (int j = j0; j < j0 + 16; ++j) {
        float s = 0.f;
        for (int k = 0; k < ZDIM; ++k) s += Cb[m * ZDIM + k] * covps[(b * ZDIM + k) * ZDIM + j];
        CP[m * ZDIM + j] = s;
      }
    }
    __syncthreads();
    // ---- S = CP @ C^T + R into augmented [S | I] ----
    {
      int m = i & 15, j0 = (i >> 4) * 8;
      for (int j = j0; j < j0 + 8; ++j) {
        float s = Rm[m * ADIM + j];
        for (int k = 0; k < ZDIM; ++k) s += CP[m * ZDIM + k] * Cb[j * ZDIM + k];
        Saug[m * 32 + j] = s;
        Saug[m * 32 + 16 + j] = (m == j) ? 1.f : 0.f;
      }
    }
    __syncthreads();
    // ---- Gauss-Jordan inverse of S (16x16) ----
    for (int p = 0; p < ADIM; ++p) {
      if (i == p) {
        float inv = 1.f / Saug[p * 32 + p];
        for (int j = 0; j < 32; ++j) Saug[p * 32 + j] *= inv;
      }
      __syncthreads();
      if (i < ADIM && i != p) {
        float f = Saug[i * 32 + p];
        for (int j = 0; j < 32; ++j) Saug[i * 32 + j] -= f * Saug[p * 32 + j];
      }
      __syncthreads();
    }
    // ---- K_t = cov_p C^T S^-1 (32x16), rows in-register ----
    {
      float cpt[ADIM];
#pragma unroll
      for (int m = 0; m < ADIM; ++m) {
        float s = 0.f;
        for (int k = 0; k < ZDIM; ++k) s += covps[(b * ZDIM + i) * ZDIM + k] * Cb[m * ZDIM + k];
        cpt[m] = s;
      }
      for (int j = 0; j < ADIM; ++j) {
        float s = 0.f;
#pragma unroll
        for (int m = 0; m < ADIM; ++m) s += cpt[m] * Saug[m * 32 + 16 + j];
        Kt[i * ADIM + j] = s;
      }
    }
    // ---- innovation = a_t - C mean_p ----
    if (i < ADIM) {
      float s = atiles[b * ADIM + i];
      for (int k = 0; k < ZDIM; ++k) s -= Cb[i * ZDIM + k] * meanps[b * ZDIM + k];
      innovs[b * ADIM + i] = s;
    }
    __syncthreads();
    // ---- mean_t ; cov_t (symmetrized closed-form, no extra barrier) ----
    {
      float s = meanps[b * ZDIM + i];
#pragma unroll
      for (int m = 0; m < ADIM; ++m) s += Kt[i * ADIM + m] * innovs[b * ADIM + m];
      meants[b * ZDIM + i] = s;
      out[o_means + brow * ZDIM + i] = s;
      for (int j = 0; j < ZDIM; ++j) {
        float t1 = 0.f, t2 = 0.f;
#pragma unroll
        for (int m = 0; m < ADIM; ++m) {
          t1 += Kt[i * ADIM + m] * CP[m * ZDIM + j];
          t2 += Kt[j * ADIM + m] * CP[m * ZDIM + i];
        }
        float cv = covps[(b * ZDIM + i) * ZDIM + j] - 0.5f * (t1 + t2);
        covt[i * ZDIM + j] = cv;
        out[o_covs + brow * ZDIM * ZDIM + i * ZDIM + j] = cv;
      }
    }
    __syncthreads();
    // ======== LSTM weight model (WMMA) ========
    if (i < ADIM) xcat0[b * K0 + i] = f2bf(atiles[b * ADIM + i]);
    __syncthreads();
    gemm_tile<KC0>(xcat0, K0, w0p, b0s, scratch, lane, wv);   // layer-0 gates
    __syncthreads();
#pragma unroll
    for (int q = 0; q < (BT * HID) / NTHREADS; ++q) {          // layer-0 cell
      int idx = tid + q * NTHREADS;
      int bb = idx >> 7, hh = idx & (HID - 1);
      const float* g = scratch + bb * SCR;
      float c = sigmoidf_(g[HID + hh]) * c0s[idx] + sigmoidf_(g[hh]) * tanhf(g[2 * HID + hh]);
      float h = sigmoidf_(g[3 * HID + hh]) * tanhf(c);
      c0s[idx] = c;
      unsigned short hb = f2bf(h);
      xcat0[bb * K0 + ADIM + hh] = hb;   // for next step's layer-0 GEMM
      xcat1[bb * K1 + hh] = hb;          // layer-1 input
    }
    __syncthreads();
    gemm_tile<KC1>(xcat1, K1, w1p, b1s, scratch, lane, wv);   // layer-1 gates
    __syncthreads();
#pragma unroll
    for (int q = 0; q < (BT * HID) / NTHREADS; ++q) {          // layer-1 cell
      int idx = tid + q * NTHREADS;
      int bb = idx >> 7, hh = idx & (HID - 1);
      const float* g = scratch + bb * SCR;
      float c = sigmoidf_(g[HID + hh]) * c1s[idx] + sigmoidf_(g[hh]) * tanhf(g[2 * HID + hh]);
      float h = sigmoidf_(g[3 * HID + hh]) * tanhf(c);
      c1s[idx] = c;
      xcat1[bb * K1 + HID + hh] = f2bf(h);
    }
    __syncthreads();
    // ---- logits = h1 @ Wout^T + bout ----
    if (tid < BT * KMIX) {
      int bb = tid >> 3, k = tid & 7;
      float s = bout[k];
      for (int j = 0; j < HID; ++j) s += bf2f(xcat1[bb * K1 + HID + j]) * Wout[k * HID + j];
      logitss[tid] = s;
    }
    __syncthreads();
    if (tid < BT) {                                            // softmax -> w_next
      float mx = -1e30f;
      for (int k = 0; k < KMIX; ++k) mx = fmaxf(mx, logitss[tid * KMIX + k]);
      float e[KMIX], sum = 0.f;
      for (int k = 0; k < KMIX; ++k) { e[k] = __expf(logitss[tid * KMIX + k] - mx); sum += e[k]; }
      float inv = 1.f / sum;
      for (int k = 0; k < KMIX; ++k) wmixs[tid * KMIX + k] = e[k] * inv;
    }
    __syncthreads();
    // ---- A_next = sum_k w_next[k] A_K[k] ----
    {
      float wr[KMIX];
#pragma unroll
      for (int k = 0; k < KMIX; ++k) wr[k] = wmixs[b * KMIX + k];
      for (int j = 0; j < ZDIM; ++j) {
        float s = 0.f;
#pragma unroll
        for (int k = 0; k < KMIX; ++k) s += wr[k] * AK[(k * ZDIM + i) * ZDIM + j];
        Ab[i * ZDIM + j] = s;
      }
    }
    __syncthreads();
    // ---- M1 = cov_t @ A_next^T ; mean_p' = A_next mean_t ----
    {
      for (int j = 0; j < ZDIM; ++j) {
        float s = 0.f;
        for (int m = 0; m < ZDIM; ++m) s += covt[i * ZDIM + m] * Ab[j * ZDIM + m];
        M1[i * ZDIM + j] = s;
      }
      float mp = 0.f;
      for (int k = 0; k < ZDIM; ++k) mp += Ab[i * ZDIM + k] * meants[b * ZDIM + k];
      meanps[b * ZDIM + i] = mp;
      out[o_nmean + brow * ZDIM + i] = mp;
    }
    __syncthreads();
    // ---- cov_p' = sym(A_next M1) + Q ----
    {
      for (int j = 0; j < ZDIM; ++j) {
        float s1 = 0.f, s2 = 0.f;
        for (int k = 0; k < ZDIM; ++k) {
          s1 += Ab[i * ZDIM + k] * M1[k * ZDIM + j];
          s2 += Ab[j * ZDIM + k] * M1[k * ZDIM + i];
        }
        float v = 0.5f * (s1 + s2) + Qm[i * ZDIM + j];
        covps[(b * ZDIM + i) * ZDIM + j] = v;
        out[o_ncov + brow * ZDIM * ZDIM + i * ZDIM + j] = v;
      }
    }
    __syncthreads();
  }

  // ---- final appended mixture matrices with w_T ----
  {
    float wr[KMIX];
#pragma unroll
    for (int k = 0; k < KMIX; ++k) wr[k] = wmixs[b * KMIX + k];
    size_t rb = (size_t)T_STEPS * BATCH + Bbase + b;
    float* oA = out + o_matA + rb * ZDIM * ZDIM + i * ZDIM;
    for (int j = 0; j < ZDIM; ++j) {
      float s = 0.f;
#pragma unroll
      for (int k = 0; k < KMIX; ++k) s += wr[k] * AK[(k * ZDIM + i) * ZDIM + j];
      oA[j] = s;
    }
    if (i < ADIM) {
      float* oC = out + o_matC + rb * ADIM * ZDIM + i * ZDIM;
      for (int j = 0; j < ZDIM; ++j) {
        float s = 0.f;
#pragma unroll
        for (int k = 0; k < KMIX; ++k) s += wr[k] * CK[(k * ADIM + i) * ZDIM + j];
        oC[j] = s;
      }
    }
  }
}

// ---------------------------------------------------------------------------
extern "C" void kernel_launch(void* const* d_in, const int* in_sizes, int n_in,
                              void* d_out, int out_size, void* d_ws, size_t ws_size,
                              hipStream_t stream)
{
  (void)in_sizes; (void)n_in; (void)out_size; (void)ws_size;
  const float* as_  = (const float*)d_in[0];
  const float* AK   = (const float*)d_in[1];
  const float* CK   = (const float*)d_in[2];
  const float* QL   = (const float*)d_in[3];
  const float* RL   = (const float*)d_in[4];
  const float* im   = (const float*)d_in[5];
  const float* ic   = (const float*)d_in[6];
  const float* Wih0 = (const float*)d_in[7];
  const float* Whh0 = (const float*)d_in[8];
  const float* bih0 = (const float*)d_in[9];
  const float* bhh0 = (const float*)d_in[10];
  const float* Wih1 = (const float*)d_in[11];
  const float* Whh1 = (const float*)d_in[12];
  const float* bih1 = (const float*)d_in[13];
  const float* bhh1 = (const float*)d_in[14];
  const float* Wout = (const float*)d_in[15];
  const float* bout = (const float*)d_in[16];

  char* ws = (char*)d_ws;
  unsigned short* w0p = (unsigned short*)(ws + WS_W0P);
  unsigned short* w1p = (unsigned short*)(ws + WS_W1P);
  float* b0s = (float*)(ws + WS_B0S);
  float* b1s = (float*)(ws + WS_B1S);
  float* Qm  = (float*)(ws + WS_QM);
  float* Rm  = (float*)(ws + WS_RM);

  const int prepN = NTILES * KC0 * 512 + NTILES * KC1 * 512
                  + 2 * G4 + ZDIM * ZDIM + ADIM * ADIM;
  ssm_prep<<<(prepN + 255) / 256, 256, 0, stream>>>(
      Wih0, Whh0, bih0, bhh0, Wih1, Whh1, bih1, bhh1, QL, RL,
      w0p, w1p, b0s, b1s, Qm, Rm);

  (void)hipFuncSetAttribute((const void*)ssm_main,
                            hipFuncAttributeMaxDynamicSharedMemorySize, SMEM_BYTES);
  ssm_main<<<NWG, NTHREADS, SMEM_BYTES, stream>>>(
      as_, AK, CK, im, ic, Wout, bout, w0p, w1p, b0s, b1s, Qm, Rm, (float*)d_out);
}
